// GATReal_4148938408768
// MI455X (gfx1250) — matrix operand order repaired and checked by reading, not compiled, not run on hardware
//
#include <hip/hip_runtime.h>

typedef _Float16 v16h __attribute__((ext_vector_type(16)));
typedef _Float16 v8h  __attribute__((ext_vector_type(8)));
typedef float    v8f  __attribute__((ext_vector_type(8)));

#define NN   10000
#define EE   80000
#define HH   40
#define AIN  128
#define EDF  8

static __device__ __forceinline__ float seluf(float x) {
    const float l = 1.0507009873554805f, a = 1.6732632423543772f;
    return x > 0.f ? l * x : l * a * (expf(x) - 1.f);
}

static __device__ __forceinline__ float atomicMaxF(float* addr, float val) {
    // IEEE ordering trick: int max for >=0, uint min for <0
    return (val >= 0.f)
        ? __int_as_float(atomicMax((int*)addr, __float_as_int(val)))
        : __uint_as_float(atomicMin((unsigned int*)addr, __float_as_uint(val)));
}

// async DMA: 16 bytes global -> LDS per lane (ASYNCcnt-tracked)
static __device__ __forceinline__ void async_g2l_b128(unsigned lds_off, const void* g) {
    asm volatile("global_load_async_to_lds_b128 %0, %1, off"
                 :: "v"(lds_off), "v"((unsigned long long)(uintptr_t)g) : "memory");
}
static __device__ __forceinline__ void wait_async0() {
    asm volatile("s_wait_asynccnt 0x0" ::: "memory");
}

// ---------------- elementwise converts / init ----------------

__global__ __launch_bounds__(256) void cvt_f32_to_f16(const float* __restrict__ in,
                                                      _Float16* __restrict__ out, int n) {
    int i = blockIdx.x * 256 + threadIdx.x;
    if (i < n) out[i] = (_Float16)in[i];
}

__global__ __launch_bounds__(256) void init_seg(float* __restrict__ nmax,
                                                float* __restrict__ nsum, int n) {
    int i = blockIdx.x * 256 + threadIdx.x;
    if (i < n) { nmax[i] = -3.4028235e38f; nsum[i] = 0.f; }
}

__global__ __launch_bounds__(256) void finalize_selu_cvt(const float* __restrict__ acc,
                                                         _Float16* __restrict__ h16, int n) {
    int i = blockIdx.x * 256 + threadIdx.x;
    if (i < n) h16[i] = (_Float16)seluf(acc[i]);
}

// ---------------- weight packing into WMMA B-fragment order ----------------
// Fragment t = tk*(Nc/16)+tn holds B tile K[tk*32..+32) x N[tn*16..+16).
// Lane L (0..31): column n = tn*16 + (L&15); K group = tk*32 + (L>>4)*16.
// Halves i=0..15 of lane L are K = Kgroup + i  (contiguous column strip).
__global__ __launch_bounds__(256) void pack_w_f16(const float* __restrict__ W,
                                                  _Float16* __restrict__ out,
                                                  int K, int Nc) {
    int idx = blockIdx.x * 256 + threadIdx.x;
    if (idx >= K * Nc) return;
    int t = idx >> 9;
    int r = idx & 511;
    int lane = r >> 4;
    int i = r & 15;
    int ntiles = Nc >> 4;
    int tk = t / ntiles;
    int tn = t - tk * ntiles;
    int k = (tk << 5) + ((lane >> 4) << 4) + i;
    int n = (tn << 4) + (lane & 15);
    out[idx] = (_Float16)W[(size_t)k * Nc + n];
}

// ---------------- WMMA GEMM:  C[M,Nc] = A16[M,K] @ W + bias ----------------
// 256 threads = 8 waves; block tile 128 x 64; wave tile 16 x 64 (4 accumulators).
// B tiles (4 KB per 32-K step) staged via async DMA into double-buffered LDS,
// shared by all 8 waves; pipeline: wait-own-async -> barrier -> issue-next ->
// batch-load all 4 B fragments -> 4 back-to-back WMMAs.
__global__ __launch_bounds__(256)
void gemm_wmma(const _Float16* __restrict__ A, const _Float16* __restrict__ PW,
               const float* __restrict__ bias, float* __restrict__ C,
               int M, int K, int Nc, int act) {
    __shared__ _Float16 BS[2][2048];            // 2 x 4KB B tiles
    const int tid  = threadIdx.x;
    const int lane = tid & 31;
    const int wave = tid >> 5;
    const int m0 = blockIdx.x * 128 + wave * 16;
    const int n0 = blockIdx.y * 64;
    const int ntiles = Nc >> 4;
    const int tn0 = n0 >> 4;
    const int S = K >> 5;                       // number of 32-deep K steps

    v8f acc[4] = {};
    int row = m0 + (lane & 15); if (row > M - 1) row = M - 1;   // clamp OOB rows
    const _Float16* arow = A + (size_t)row * K + ((lane >> 4) << 4);

    // prologue: stage first B tile (256 threads x 16B = 4KB)
    async_g2l_b128((unsigned)(uintptr_t)&BS[0][tid * 8],
                   PW + (size_t)tn0 * 512 + tid * 8);

    for (int s = 0; s < S; ++s) {
        wait_async0();                          // my slice of tile s done
        __syncthreads();                        // everyone's slices done
        if (s + 1 < S) {                        // DMA next tile into other buffer
            async_g2l_b128((unsigned)(uintptr_t)&BS[(s + 1) & 1][tid * 8],
                           PW + ((size_t)(s + 1) * ntiles + tn0) * 512 + tid * 8);
        }
        const int k0 = s << 5;
        v8h lo = *(const v8h*)(arow + k0);
        v8h hi = *(const v8h*)(arow + k0 + 8);
        if (k0 + 32 < K) __builtin_prefetch(arow + k0 + 32, 0, 1);
        v16h a;
#pragma unroll
        for (int i = 0; i < 8; ++i) { a[i] = lo[i]; a[i + 8] = hi[i]; }

        // batch all LDS fragment loads first, then run the WMMA chain
        const _Float16* bb = &BS[s & 1][lane * 16];
        v16h b[4];
#pragma unroll
        for (int j = 0; j < 4; ++j) b[j] = *(const v16h*)(bb + (j << 9));
#pragma unroll
        for (int j = 0; j < 4; ++j) {
            acc[j] = __builtin_amdgcn_wmma_f32_16x16x32_f16(
                false, a, false, b[j], (short)0, acc[j], false, false);
        }
    }

    const int rbase = m0 + ((lane >> 4) << 3);
    const int cb = lane & 15;
#pragma unroll
    for (int j = 0; j < 4; ++j) {
        int col = n0 + j * 16 + cb;
        float bv = bias ? bias[col] : 0.f;
#pragma unroll
        for (int r = 0; r < 8; ++r) {
            int rr = rbase + r;
            if (rr < M) {
                float v = acc[j][r] + bv;
                if (act == 1) v = seluf(v);
                C[(size_t)rr * Nc + col] = v;
            }
        }
    }
}

// ---------------- GATv2 edge pipeline ----------------

// One wave per (edge, head): logit[e,h] = sum_j relu(xl[s]+xr[d]+ea@We)[h,j]*att[h,j]
__global__ __launch_bounds__(256)
void edge_logit(const float* __restrict__ xl, const float* __restrict__ xr,
                const float* __restrict__ ea, const float* __restrict__ We,
                const float* __restrict__ att, const int* __restrict__ src,
                const int* __restrict__ dst, float* __restrict__ logit,
                int HC, int c) {
    int w = blockIdx.x * 8 + (threadIdx.x >> 5);
    int lane = threadIdx.x & 31;
    if (w >= EE * HH) return;                  // wave-uniform
    int e = w / HH, h = w - e * HH;
    int s = src[e], d = dst[e];
    float ev[EDF];
#pragma unroll
    for (int t = 0; t < EDF; ++t) ev[t] = ea[(size_t)e * EDF + t];
    float val = 0.f;
    for (int j = lane; j < c; j += 32) {
        int idx = h * c + j;
        float m = xl[(size_t)s * HC + idx] + xr[(size_t)d * HC + idx];
#pragma unroll
        for (int t = 0; t < EDF; ++t) m += ev[t] * We[(size_t)t * HC + idx];
        m = m > 0.f ? m : 0.f;
        val += m * att[idx];
    }
    for (int off = 16; off; off >>= 1) val += __shfl_xor(val, off, 32);
    if (lane == 0) logit[(size_t)e * HH + h] = val;
}

__global__ __launch_bounds__(256)
void edge_max(const float* __restrict__ logit, const int* __restrict__ dst,
              float* __restrict__ nmax) {
    int i = blockIdx.x * 256 + threadIdx.x;
    if (i >= EE * HH) return;
    int e = i / HH, h = i - e * HH;
    atomicMaxF(&nmax[(size_t)dst[e] * HH + h], logit[i]);
}

__global__ __launch_bounds__(256)
void edge_expsum(const float* __restrict__ logit, const int* __restrict__ dst,
                 const float* __restrict__ nmax, float* __restrict__ ex,
                 float* __restrict__ nsum) {
    int i = blockIdx.x * 256 + threadIdx.x;
    if (i >= EE * HH) return;
    int e = i / HH, h = i - e * HH;
    int d = dst[e];
    float v = expf(logit[i] - nmax[(size_t)d * HH + h]);
    ex[i] = v;
    atomicAdd(&nsum[(size_t)d * HH + h], v);
}

// One block per edge: acc[dst] += alpha * xl[src]   (acc pre-loaded with residual)
__global__ __launch_bounds__(256)
void edge_scatter(const float* __restrict__ ex, const float* __restrict__ nsum,
                  const float* __restrict__ xl, const int* __restrict__ src,
                  const int* __restrict__ dst, float* __restrict__ acc,
                  int HC, int lc) {
    int e = blockIdx.x;
    int s = src[e], d = dst[e];
    for (int idx = threadIdx.x; idx < HC; idx += 256) {
        int h = idx >> lc;
        float a = ex[(size_t)e * HH + h] / (nsum[(size_t)d * HH + h] + 1e-16f);
        atomicAdd(&acc[(size_t)d * HC + idx], a * xl[(size_t)s * HC + idx]);
    }
}

// ---------------- batch norm ----------------

__global__ __launch_bounds__(256)
void bn_stats(const float* __restrict__ X, int M, int C,
              float* __restrict__ mu, float* __restrict__ rv) {
    __shared__ float s1[256], s2[256];
    int col = blockIdx.x, tid = threadIdx.x;
    float a = 0.f, b = 0.f;
    for (int r = tid; r < M; r += 256) {
        float v = X[(size_t)r * C + col];
        a += v; b += v * v;
    }
    s1[tid] = a; s2[tid] = b; __syncthreads();
    for (int o = 128; o; o >>= 1) {
        if (tid < o) { s1[tid] += s1[tid + o]; s2[tid] += s2[tid + o]; }
        __syncthreads();
    }
    if (tid == 0) {
        float m = s1[0] / (float)M;
        float var = s2[0] / (float)M - m * m;
        mu[col] = m;
        rv[col] = rsqrtf(var + 1e-5f);
    }
}

__global__ __launch_bounds__(256)
void bn_apply_selu_cvt(const float* __restrict__ X, const float* __restrict__ mu,
                       const float* __restrict__ rv, const float* __restrict__ g,
                       const float* __restrict__ b, _Float16* __restrict__ out,
                       int M, int C) {
    int i = blockIdx.x * 256 + threadIdx.x;
    if (i >= M * C) return;
    int col = i % C;
    float v = (X[i] - mu[col]) * rv[col] * g[col] + b[col];
    out[i] = (_Float16)seluf(v);
}

// ---------------- tiny heads (lin4: 512->4, lin5: 512->1) ----------------

__global__ __launch_bounds__(256)
void small_heads(const _Float16* __restrict__ h, const float* __restrict__ W4,
                 const float* __restrict__ b4, const float* __restrict__ W5,
                 const float* __restrict__ b5, float* __restrict__ bb,
                 float* __restrict__ p, int M) {
    int r = blockIdx.x * 256 + threadIdx.x;
    if (r >= M) return;
    float a0 = b4[0], a1 = b4[1], a2 = b4[2], a3 = b4[3], a5 = b5[0];
    const _Float16* hr = h + (size_t)r * 512;
    for (int k = 0; k < 512; ++k) {
        float v = (float)hr[k];
        a0 += v * W4[k * 4 + 0];
        a1 += v * W4[k * 4 + 1];
        a2 += v * W4[k * 4 + 2];
        a3 += v * W4[k * 4 + 3];
        a5 += v * W5[k];
    }
    bb[(size_t)r * 4 + 0] = a0; bb[(size_t)r * 4 + 1] = a1;
    bb[(size_t)r * 4 + 2] = a2; bb[(size_t)r * 4 + 3] = a3;
    p[r] = a5;
}

// ---------------- host orchestration ----------------

extern "C" void kernel_launch(void* const* d_in, const int* in_sizes, int n_in,
                              void* d_out, int out_size, void* d_ws, size_t ws_size,
                              hipStream_t stream) {
    const float* x  = (const float*)d_in[0];
    const int*   ei = (const int*)d_in[1];
    const float* ea = (const float*)d_in[2];
    const int* src = ei;
    const int* dst = ei + EE;

    // GAT layer params: [Wl,bl,Wr,br,We,att,Wres,b] starting at index 3, 8 each
    const float* G[3][8];
    for (int L = 0; L < 3; ++L)
        for (int k = 0; k < 8; ++k) G[L][k] = (const float*)d_in[3 + L * 8 + k];
    const float* lin1_W = (const float*)d_in[27]; const float* lin1_b = (const float*)d_in[28];
    const float* bn1_g  = (const float*)d_in[29]; const float* bn1_b  = (const float*)d_in[30];
    const float* lin2_W = (const float*)d_in[31]; const float* lin2_b = (const float*)d_in[32];
    const float* bn2_g  = (const float*)d_in[33]; const float* bn2_b  = (const float*)d_in[34];
    const float* lin3_W = (const float*)d_in[35]; const float* lin3_b = (const float*)d_in[36];
    const float* lin4_W = (const float*)d_in[37]; const float* lin4_b = (const float*)d_in[38];
    const float* lin5_W = (const float*)d_in[39]; const float* lin5_b = (const float*)d_in[40];

    float* out = (float*)d_out;

    // workspace layout
    char* base = (char*)d_ws;
    const size_t HC3 = 2560;
    float*    XL    = (float*)(base);                                   // N*2560 f32
    float*    XRACC = (float*)(base + (size_t)NN * HC3 * 4);            // N*2560 f32
    _Float16* H16   = (_Float16*)(base + (size_t)NN * HC3 * 8);         // N*2560 f16
    _Float16* PW    = (_Float16*)(base + (size_t)NN * HC3 * 10);        // packed weights
    char* p2 = base + (size_t)NN * HC3 * 10 + 8ull * 1024 * 1024;
    float* LOGIT = (float*)p2;                                          // E*H
    float* EX    = LOGIT + (size_t)EE * HH;
    float* NMAX  = EX + (size_t)EE * HH;
    float* NSUM  = NMAX + (size_t)NN * HH;
    float* MU    = NSUM + (size_t)NN * HH;
    float* RV    = MU + 1024;

    auto grid1 = [](int n) { return dim3((n + 255) / 256); };

    auto pack = [&](const float* W, int K, int Nc) {
        pack_w_f16<<<grid1(K * Nc), 256, 0, stream>>>(W, PW, K, Nc);
    };
    auto gemm = [&](const _Float16* A, const float* bias, float* C,
                    int M, int K, int Nc, int act) {
        dim3 g((M + 127) / 128, Nc / 64);
        gemm_wmma<<<g, 256, 0, stream>>>(A, PW, bias, C, M, K, Nc, act);
    };

    // x -> f16
    cvt_f32_to_f16<<<grid1(NN * AIN), 256, 0, stream>>>(x, H16, NN * AIN);

    const int cs[3] = {16, 32, 64}, lcs[3] = {4, 5, 6};
    const int Ks[3] = {AIN, 640, 1280}, HCs[3] = {640, 1280, 2560};

    for (int L = 0; L < 3; ++L) {
        int K = Ks[L], c = cs[L], HC = HCs[L], lc = lcs[L];
        const float *Wl = G[L][0], *bl = G[L][1], *Wr = G[L][2], *br = G[L][3];
        const float *We = G[L][4], *att = G[L][5], *Wres = G[L][6], *bres = G[L][7];

        pack(Wl, K, HC); gemm(H16, bl, XL, NN, K, HC, 0);        // xl
        pack(Wr, K, HC); gemm(H16, br, XRACC, NN, K, HC, 0);     // xr
        init_seg<<<grid1(NN * HH), 256, 0, stream>>>(NMAX, NSUM, NN * HH);
        edge_logit<<<(EE * HH + 7) / 8, 256, 0, stream>>>(XL, XRACC, ea, We, att,
                                                          src, dst, LOGIT, HC, c);
        edge_max<<<grid1(EE * HH), 256, 0, stream>>>(LOGIT, dst, NMAX);
        edge_expsum<<<grid1(EE * HH), 256, 0, stream>>>(LOGIT, dst, NMAX, EX, NSUM);
        pack(Wres, K, HC); gemm(H16, bres, XRACC, NN, K, HC, 0); // acc := residual
        edge_scatter<<<EE, 256, 0, stream>>>(EX, NSUM, XL, src, dst, XRACC, HC, lc);
        finalize_selu_cvt<<<grid1(NN * HC), 256, 0, stream>>>(XRACC, H16, NN * HC);
    }

    // MLP head
    pack(lin1_W, 2560, 1024); gemm(H16, lin1_b, XL, NN, 2560, 1024, 0);
    bn_stats<<<1024, 256, 0, stream>>>(XL, NN, 1024, MU, RV);
    bn_apply_selu_cvt<<<grid1(NN * 1024), 256, 0, stream>>>(XL, MU, RV, bn1_g, bn1_b,
                                                            H16, NN, 1024);
    pack(lin2_W, 1024, 512); gemm(H16, lin2_b, XRACC, NN, 1024, 512, 0);
    bn_stats<<<512, 256, 0, stream>>>(XRACC, NN, 512, MU, RV);
    bn_apply_selu_cvt<<<grid1(NN * 512), 256, 0, stream>>>(XRACC, MU, RV, bn2_g, bn2_b,
                                                           H16, NN, 512);
    pack(lin3_W, 512, 64); gemm(H16, lin3_b, out, NN, 512, 64, 0);   // RF
    small_heads<<<grid1(NN), 256, 0, stream>>>(H16, lin4_W, lin4_b, lin5_W, lin5_b,
                                               out + (size_t)NN * 64,
                                               out + (size_t)NN * 68, NN);
}
